// Model_62904091018076
// MI455X (gfx1250) — compile-verified, matrix operand order checked by reading, not run on hardware
//
#include <hip/hip_runtime.h>

#define BATCH   256
#define SEQL    20000
#define WPOOL   3999
#define NHEADS  5
#define HDIM    30
#define NQKV    450      // 3 * 5 * 30 (q,k,v fused along N)
#define NTILES  32       // 32 tiles * 16 = 512 padded N
#define KSPAD   128      // padded K-steps (125 real, 3 zero)
#define XSTRIDE 4096     // padded activation row stride (halfs) = KSPAD*32

typedef __attribute__((ext_vector_type(16))) _Float16 v16h;
typedef __attribute__((ext_vector_type(8)))  float    v8f;
typedef int v4i __attribute__((vector_size(16)));
typedef __attribute__((address_space(1))) v4i* gv4p;   // global AS pointer
typedef __attribute__((address_space(3))) v4i* lv4p;   // LDS AS pointer

// Async global->LDS staging if the toolchain declares the gfx1250 builtins.
#if defined(__has_builtin)
#if __has_builtin(__builtin_amdgcn_global_load_async_to_lds_b128) && \
    __has_builtin(__builtin_amdgcn_s_wait_asynccnt)
#define USE_ASYNC_LDS 1
#endif
#endif

static __device__ __forceinline__ int imin(int a, int b) { return a < b ? a : b; }
static __device__ __forceinline__ int imax(int a, int b) { return a > b ? a : b; }

// ---------------------------------------------------------------------------
// Kernel 0a: collapse conv1(one-hot) + conv2 into lookup tables.
//   U[dh2][dw2][dh1][dw1] = sum_c w1[dh1,dw1,0,c] * w2[dh2,dw2,c,0]   (1600 f)
//   Bw2[dh2][dw2]         = sum_c b1[c] * w2[dh2,dw2,c,0]             (40 f)
//   T[d][t][i]            = composite tap table, d=dw1+dw2 in [0,18]  (304 f)
//   biasInt[i]            = b2 + interior bias                        (4 f)
// ---------------------------------------------------------------------------
__global__ void __launch_bounds__(256) prep_tables(
    const float* __restrict__ w1, const float* __restrict__ b1,
    const float* __restrict__ w2, const float* __restrict__ b2,
    float* __restrict__ U, float* __restrict__ Bw2,
    float* __restrict__ biasInt, float* __restrict__ T)
{
    __shared__ float Ul[1600];
    __shared__ float Bl[40];
    int tid = threadIdx.x;

    for (int idx = tid; idx < 1600; idx += 256) {
        int dw1 = idx % 10, dh1 = (idx / 10) % 4, dw2 = (idx / 40) % 10, dh2 = idx / 400;
        float s = 0.f;
        #pragma unroll
        for (int c = 0; c < 5; ++c)
            s += w1[(dh1 * 10 + dw1) * 5 + c] * w2[(dh2 * 10 + dw2) * 5 + c];
        Ul[idx] = s;
        U[idx]  = s;
    }
    for (int idx = tid; idx < 40; idx += 256) {
        int dh2 = idx / 10, dw2 = idx % 10;
        float s = 0.f;
        #pragma unroll
        for (int c = 0; c < 5; ++c) s += b1[c] * w2[(dh2 * 10 + dw2) * 5 + c];
        Bl[idx]  = s;
        Bw2[idx] = s;
    }
    __syncthreads();

    for (int idx = tid; idx < 304; idx += 256) {
        int i = idx & 3, t = (idx >> 2) & 3, d = idx >> 4;   // d in [0,18]
        float s = 0.f;
        for (int dh2 = 0; dh2 < 4; ++dh2) {
            int r = i + dh2 - 1;             // conv1 output row
            if (r < 0 || r > 3) continue;
            int dh1 = t - r + 1;             // one-hot row hit inside conv1 kernel
            if (dh1 < 0 || dh1 > 3) continue;
            int lo = imax(0, d - 9), hi = imin(9, d);
            for (int dw2 = lo; dw2 <= hi; ++dw2) {
                int dw1 = d - dw2;
                s += Ul[((dh2 * 10 + dw2) * 4 + dh1) * 10 + dw1];
            }
        }
        T[idx] = s;
    }
    if (tid < 4) {
        float s = b2[0];
        for (int dh2 = 0; dh2 < 4; ++dh2) {
            int r = tid + dh2 - 1;
            if (r < 0 || r > 3) continue;
            for (int dw2 = 0; dw2 < 10; ++dw2) s += Bl[dh2 * 10 + dw2];
        }
        biasInt[tid] = s;
    }
}

// ---------------------------------------------------------------------------
// Kernel 0b: convert Wq/Wk/Wv (fp32) into the WMMA B-operand per-lane swizzle
// (f16).  Layout: Bsw[ntile][kstep][lane][16 halfs], matching ISA 7.12.2:
//   lane<16  -> N = ntile*16+lane,    K in {k0+0..7,  k0+16..23}
//   lane>=16 -> N = ntile*16+lane-16, K in {k0+8..15, k0+24..31}
// kstep padded to 128 (k >= 3999 zero-filled).
// ---------------------------------------------------------------------------
__global__ void __launch_bounds__(256) prep_weights(
    const float* __restrict__ Wq, const float* __restrict__ Wk,
    const float* __restrict__ Wv, _Float16* __restrict__ Bsw)
{
    const long TOT = (long)NTILES * KSPAD * 32 * 16;     // 2,097,152
    for (long idx = (long)blockIdx.x * blockDim.x + threadIdx.x; idx < TOT;
         idx += (long)gridDim.x * blockDim.x) {
        int e     = (int)(idx & 15);
        int lane  = (int)((idx >> 4) & 31);
        int kstep = (int)((idx >> 9) & (KSPAD - 1));
        int ntile = (int)(idx >> 16);
        int base8 = (lane >= 16) ? 8 : 0;
        int n     = ntile * 16 + (lane & 15);
        int k     = kstep * 32 + base8 + (e < 8 ? e : e + 8);
        float val = 0.f;
        if (n < NQKV && k < WPOOL) {
            int m  = n / 150;
            int hh = (n % 150) / 30;
            int ee = n % 30;
            const float* W = (m == 0) ? Wq : ((m == 1) ? Wk : Wv);
            val = W[((long)hh * WPOOL + k) * HDIM + ee];
        }
        Bsw[idx] = (_Float16)val;
    }
}

// ---------------------------------------------------------------------------
// Kernel 1: fused conv1+conv2 (as table lookups) + maxpool(10, stride 5).
// grid = (16 pool tiles, 256 batches), block = 256.
// Interior columns: 19 taps * float4 LDS gather. Edge columns (~18/20000):
// exact nested-loop path through U/Bw2.
// Output x as f16 with padded row stride: xh[(b*4 + row)*XSTRIDE + p].
// ---------------------------------------------------------------------------
__global__ void __launch_bounds__(256) conv_pool(
    const int* __restrict__ tok, const float* __restrict__ U,
    const float* __restrict__ Bw2, const float* __restrict__ biasInt,
    const float* __restrict__ T, const float* __restrict__ b2,
    _Float16* __restrict__ xh)
{
    __shared__ __align__(16) float Tl[304];
    __shared__ int   tokl[1312];
    __shared__ float convl[4 * 1288];

    const int tid  = threadIdx.x;
    const int b    = blockIdx.y;
    const int p0   = blockIdx.x * 256;
    const int pEnd = imin(p0 + 256, WPOOL);
    const int npool = pEnd - p0;
    const int c0   = p0 * 5;
    const int ncols = (pEnd - 1) * 5 + 10 - c0;
    const int ntok  = ncols + 18;

    __builtin_prefetch(&tok[(long)b * SEQL + imax(0, c0 - 8)], 0, 0);

    for (int i = tid; i < 304; i += 256) Tl[i] = T[i];
    for (int i = tid; i < ntok; i += 256) {
        int g = c0 - 8 + i;
        tokl[i] = (g >= 0 && g < SEQL) ? tok[(long)b * SEQL + g] : 0;
    }
    const float bi0 = biasInt[0], bi1 = biasInt[1], bi2 = biasInt[2], bi3 = biasInt[3];
    __syncthreads();

    for (int c = tid; c < ncols; c += 256) {
        int j = c0 + c;
        float a0, a1, a2, a3;
        if (j >= 8 && j <= SEQL - 11) {
            a0 = bi0; a1 = bi1; a2 = bi2; a3 = bi3;
            #pragma unroll
            for (int d = 0; d < 19; ++d) {
                int t = tokl[c + d];
                const float4 tv = *reinterpret_cast<const float4*>(&Tl[((d << 2) + t) << 2]);
                a0 += tv.x; a1 += tv.y; a2 += tv.z; a3 += tv.w;
            }
        } else {
            float acc[4];
            float b2v = b2[0];
            acc[0] = b2v; acc[1] = b2v; acc[2] = b2v; acc[3] = b2v;
            for (int dh2 = 0; dh2 < 4; ++dh2) {
                for (int dw2 = 0; dw2 < 10; ++dw2) {
                    int s = j + dw2 - 4;
                    if (s < 0 || s >= SEQL) continue;
                    float bwc = Bw2[dh2 * 10 + dw2];
                    #pragma unroll
                    for (int ii = 0; ii < 4; ++ii) {
                        int r = ii + dh2 - 1;
                        if (r >= 0 && r <= 3) acc[ii] += bwc;
                    }
                    for (int dw1 = 0; dw1 < 10; ++dw1) {
                        int w = s + dw1 - 4;
                        if (w < 0 || w >= SEQL) continue;
                        int t = tokl[c + 8 + (w - j)];
                        #pragma unroll
                        for (int ii = 0; ii < 4; ++ii) {
                            int r = ii + dh2 - 1;
                            if (r < 0 || r > 3) continue;
                            int dh1 = t - r + 1;
                            if (dh1 < 0 || dh1 > 3) continue;
                            acc[ii] += U[((dh2 * 10 + dw2) * 4 + dh1) * 10 + dw1];
                        }
                    }
                }
            }
            a0 = acc[0]; a1 = acc[1]; a2 = acc[2]; a3 = acc[3];
        }
        convl[0 * 1288 + c] = a0;
        convl[1 * 1288 + c] = a1;
        convl[2 * 1288 + c] = a2;
        convl[3 * 1288 + c] = a3;
    }
    __syncthreads();

    for (int pi = tid; pi < npool; pi += 256) {
        int p = p0 + pi;
        int base = pi * 5;
        #pragma unroll
        for (int i = 0; i < 4; ++i) {
            float m = convl[i * 1288 + base];
            #pragma unroll
            for (int o = 1; o < 10; ++o) {
                float v = convl[i * 1288 + base + o];
                m = (v > m) ? v : m;
            }
            xh[((long)b * 4 + i) * XSTRIDE + p] = (_Float16)m;
        }
    }

    // Zero-fill the K padding columns [WPOOL, XSTRIDE) once per batch.
    if (blockIdx.x == 15) {
        const int npad = XSTRIDE - WPOOL;               // 97
        for (int idx = tid; idx < npad * 4; idx += 256) {
            int i = idx & 3;
            int p = WPOOL + (idx >> 2);
            xh[((long)b * 4 + i) * XSTRIDE + p] = (_Float16)0.f;
        }
    }
}

// ---------------------------------------------------------------------------
// Kernel 2: QKV projection GEMM with v_wmma_f32_16x16x32_f16.
// M = 1024 (4 batches * 4 rows per 16-row tile), K = 4096 padded, N = 450
// (padded 512). 64 blocks (one per M-tile) * 8 waves; each wave owns 4
// N-tiles. A staged 128 K at a time (one b128 per thread, async-to-LDS when
// available); B pre-swizzled per lane (one 32B vector load per WMMA).
// ---------------------------------------------------------------------------
__global__ void __launch_bounds__(256) qkv_gemm(
    const _Float16* __restrict__ xh, const _Float16* __restrict__ Bsw,
    float* __restrict__ qkv)
{
    __shared__ __align__(16) _Float16 As[16 * 128];     // 4 KB: 16 rows x 128 K

    const int tid   = threadIdx.x;
    const int lane  = tid & 31;
    const int wave  = tid >> 5;
    const int btile = blockIdx.x;           // 0..63

    v8f acc[4] = {};
    const int row   = lane & 15;
    const int base8 = (lane & 16) ? 8 : 0;
    const v16h* __restrict__ bp = reinterpret_cast<const v16h*>(Bsw);

    // Per-thread staging coordinates: 2048 halfs / 256 threads = 8 halfs (16B).
    const int sidx = tid * 8;
    const int srow = sidx >> 7;             // 0..15
    const int skk  = sidx & 127;
    const int sbb  = btile * 4 + (srow >> 2);
    const int sii  = srow & 3;
    const _Float16* sptr = xh + ((long)sbb * 4 + sii) * XSTRIDE + skk;

    for (int st = 0; st < KSPAD / 4; ++st) {            // 32 stages of 128 K
#if defined(USE_ASYNC_LDS)
        {
            gv4p gp = (gv4p)(unsigned long long)(const void*)(sptr + st * 128);
            lv4p lp = (lv4p)(unsigned int)(unsigned long long)(void*)&As[sidx];
            __builtin_amdgcn_global_load_async_to_lds_b128(gp, lp, 0, 0);
            __builtin_amdgcn_s_wait_asynccnt(0);
        }
#else
        *reinterpret_cast<uint4*>(&As[sidx]) =
            *reinterpret_cast<const uint4*>(sptr + st * 128);
#endif
        __syncthreads();

        #pragma unroll
        for (int ksub = 0; ksub < 4; ++ksub) {
            v16h a;
            #pragma unroll
            for (int e = 0; e < 16; ++e) {
                int k = ksub * 32 + base8 + (e < 8 ? e : e + 8);
                a[e] = As[row * 128 + k];
            }
            const int ks = st * 4 + ksub;
            #pragma unroll
            for (int nt = 0; nt < 4; ++nt) {
                int ntile = wave * 4 + nt;
                v16h bm = bp[((long)ntile * KSPAD + ks) * 32 + lane];
                acc[nt] = __builtin_amdgcn_wmma_f32_16x16x32_f16(
                    false, a, false, bm, (short)0, acc[nt], false, false);
            }
        }
        __syncthreads();
    }

    // C layout (ISA 7.12.2): VGPR r -> lanes0-15 M=r, lanes16-31 M=r+8; N = lane%16.
    const int nloc    = lane & 15;
    const int rowBase = (lane & 16) ? 8 : 0;
    for (int nt = 0; nt < 4; ++nt) {
        int n = (wave * 4 + nt) * 16 + nloc;
        if (n < NQKV) {
            int m  = n / 150;
            int hh = (n % 150) / 30;
            int ee = n % 30;
            #pragma unroll
            for (int r = 0; r < 8; ++r) {
                int rowM = r + rowBase;
                int bb = btile * 4 + (rowM >> 2);
                int ss = rowM & 3;
                qkv[((((long)bb * 3 + m) * NHEADS + hh) * 4 + ss) * HDIM + ee] = acc[nt][r];
            }
        }
    }
}

// ---------------------------------------------------------------------------
// Kernel 3: causal attention over S=4, Wo projection, MLP, sigmoid.
// One block (160 threads) per batch; everything cooperative through LDS.
// ---------------------------------------------------------------------------
__global__ void __launch_bounds__(160) attn_mlp(
    const float* __restrict__ qkv, const float* __restrict__ Wo,
    const float* __restrict__ d1w, const float* __restrict__ d1b,
    const float* __restrict__ d2w, const float* __restrict__ d2b,
    const float* __restrict__ d3w, const float* __restrict__ d3b,
    float* __restrict__ out)
{
    __shared__ float sc[NHEADS * 4 * 4];
    __shared__ float ol[4 * 150];
    __shared__ float hf[120];
    __shared__ float h1[10];
    __shared__ float h2[10];

    const int b   = blockIdx.x;
    const int tid = threadIdx.x;
    const float* q  = qkv + (((long)b * 3 + 0) * NHEADS) * 4 * HDIM;
    const float* kM = qkv + (((long)b * 3 + 1) * NHEADS) * 4 * HDIM;
    const float* v  = qkv + (((long)b * 3 + 2) * NHEADS) * 4 * HDIM;

    if (tid < 80) {                                     // scores
        int h = tid / 16, s = (tid >> 2) & 3, t = tid & 3;
        float r = 0.f;
        if (t <= s) {
            const float* qp = q  + (h * 4 + s) * HDIM;
            const float* kp = kM + (h * 4 + t) * HDIM;
            #pragma unroll
            for (int e = 0; e < HDIM; ++e) r += qp[e] * kp[e];
            r *= 0.18257418583505536f;                  // 1/sqrt(30)
        }
        sc[tid] = r;
    }
    __syncthreads();

    if (tid < 20) {                                     // causal softmax
        int h = tid / 4, s = tid % 4;
        float mx = -1e30f;
        for (int t = 0; t <= s; ++t) mx = fmaxf(mx, sc[(h * 4 + s) * 4 + t]);
        float ex[4]; float sum = 0.f;
        #pragma unroll
        for (int t = 0; t < 4; ++t) {
            ex[t] = (t <= s) ? __expf(sc[(h * 4 + s) * 4 + t] - mx) : 0.f;
            sum += ex[t];
        }
        float inv = 1.f / sum;
        #pragma unroll
        for (int t = 0; t < 4; ++t) sc[(h * 4 + s) * 4 + t] = ex[t] * inv;
    }
    __syncthreads();

    if (tid < 150) {                                    // o = attn @ v
        int h = tid / HDIM, e = tid % HDIM;
        #pragma unroll
        for (int s = 0; s < 4; ++s) {
            float r = 0.f;
            for (int t = 0; t <= s; ++t)
                r += sc[(h * 4 + s) * 4 + t] * v[(h * 4 + t) * HDIM + e];
            ol[s * 150 + h * HDIM + e] = r;
        }
    }
    __syncthreads();

    if (tid < 120) {                                    // o @ Wo, flatten
        int s = tid / HDIM, e2 = tid % HDIM;
        float r = 0.f;
        for (int k = 0; k < 150; ++k) r += ol[s * 150 + k] * Wo[k * HDIM + e2];
        hf[s * HDIM + e2] = r;
    }
    __syncthreads();

    if (tid < 10) {                                     // dense1 + leaky relu
        float r = d1b[tid];
        for (int k = 0; k < 120; ++k) r += hf[k] * d1w[k * 10 + tid];
        h1[tid] = (r > 0.f) ? r : 0.2f * r;
    }
    __syncthreads();
    if (tid < 10) {                                     // dense2 + leaky relu
        float r = d2b[tid];
        for (int k = 0; k < 10; ++k) r += h1[k] * d2w[k * 10 + tid];
        h2[tid] = (r > 0.f) ? r : 0.2f * r;
    }
    __syncthreads();
    if (tid == 0) {                                     // dense3 + sigmoid
        float r = d3b[0];
        for (int k = 0; k < 10; ++k) r += h2[k] * d3w[k];
        out[b] = 1.f / (1.f + __expf(-r));
    }
}

// ---------------------------------------------------------------------------
extern "C" void kernel_launch(void* const* d_in, const int* in_sizes, int n_in,
                              void* d_out, int out_size, void* d_ws, size_t ws_size,
                              hipStream_t stream)
{
    (void)in_sizes; (void)n_in; (void)out_size; (void)ws_size;
    const int*   tokens = (const int*)  d_in[0];
    const float* c1w    = (const float*)d_in[1];
    const float* c1b    = (const float*)d_in[2];
    const float* c2w    = (const float*)d_in[3];
    const float* c2b    = (const float*)d_in[4];
    const float* Wq     = (const float*)d_in[5];
    const float* Wk     = (const float*)d_in[6];
    const float* Wv     = (const float*)d_in[7];
    const float* Wo     = (const float*)d_in[8];
    const float* d1w    = (const float*)d_in[9];
    const float* d1b    = (const float*)d_in[10];
    const float* d2w    = (const float*)d_in[11];
    const float* d2b    = (const float*)d_in[12];
    const float* d3w    = (const float*)d_in[13];
    const float* d3b    = (const float*)d_in[14];

    char* base = (char*)d_ws;
    size_t off = 0;
    auto carve = [&](size_t bytes) -> char* {
        char* p = base + off;
        off = (off + bytes + 255) & ~(size_t)255;
        return p;
    };
    float*    U   = (float*)   carve(1600 * sizeof(float));
    float*    Bw2 = (float*)   carve(40 * sizeof(float));
    float*    bia = (float*)   carve(4 * sizeof(float));
    float*    T   = (float*)   carve(304 * sizeof(float));
    _Float16* Bsw = (_Float16*)carve((size_t)NTILES * KSPAD * 32 * 16 * 2);
    _Float16* xh  = (_Float16*)carve((size_t)BATCH * 4 * XSTRIDE * 2);
    float*    qkv = (float*)   carve((size_t)BATCH * 3 * NHEADS * 4 * HDIM * sizeof(float));

    prep_tables<<<1, 256, 0, stream>>>(c1w, c1b, c2w, c2b, U, Bw2, bia, T);
    prep_weights<<<2048, 256, 0, stream>>>(Wq, Wk, Wv, Bsw);
    conv_pool<<<dim3(16, BATCH), 256, 0, stream>>>(tokens, U, Bw2, bia, T, c2b, xh);
    qkv_gemm<<<64, 256, 0, stream>>>(xh, Bsw, qkv);
    attn_mlp<<<BATCH, 160, 0, stream>>>(qkv, Wo, d1w, d1b, d2w, d2b, d3w, d3b,
                                        (float*)d_out);
}
